// MambaBlock_42752104465111
// MI455X (gfx1250) — matrix operand order repaired and checked by reading, not compiled, not run on hardware
//
#include <hip/hip_runtime.h>
#include <hip/hip_bf16.h>

// ---------------------------------------------------------------------------
// MambaBlock for MI455X (gfx1250): wave32 + WMMA f16->f32 GEMMs with
// async-to-LDS double-buffered staging; fp32 selective scan.
// ---------------------------------------------------------------------------

typedef __attribute__((ext_vector_type(16))) _Float16 v16h;
typedef __attribute__((ext_vector_type(8)))  float    v8f;

#define D_MODEL 1024
#define D_INNER 2048
#define D_STATE 2
#define D_CONV  4
#define DT_RANK 64
#define XPROJ_N (DT_RANK + 2 * D_STATE)   // 68
#define BATCH   2
#define SEQ     4096
#define TOKENS  (BATCH * SEQ)             // 8192

// ------------------------------ fp32 -> fp16 -------------------------------
__global__ void cvt_f32_to_f16(const float* __restrict__ in,
                               _Float16* __restrict__ out, int n) {
  int i = blockIdx.x * blockDim.x + threadIdx.x;
  if (i < n) out[i] = (_Float16)in[i];
}

// ------------------------------- LayerNorm ---------------------------------
// One wave32 per row of 1024; 4 rows per 128-thread block. fp16 output.
__global__ void layernorm_f16(const float* __restrict__ x,
                              const float* __restrict__ g,
                              const float* __restrict__ b,
                              _Float16* __restrict__ out) {
  const int row  = blockIdx.x * 4 + (threadIdx.x >> 5);
  const int lane = threadIdx.x & 31;
  const float* xr = x + (size_t)row * D_MODEL;

  float vals[D_MODEL / 32];
  float s = 0.f;
#pragma unroll
  for (int i = 0; i < D_MODEL / 32; ++i) {
    float v = xr[lane + i * 32];
    vals[i] = v;
    s += v;
  }
#pragma unroll
  for (int m = 16; m; m >>= 1) s += __shfl_xor(s, m, 32);
  const float mu = s * (1.f / D_MODEL);

  float vs = 0.f;
#pragma unroll
  for (int i = 0; i < D_MODEL / 32; ++i) {
    float d = vals[i] - mu;
    vs += d * d;
  }
#pragma unroll
  for (int m = 16; m; m >>= 1) vs += __shfl_xor(vs, m, 32);
  const float rinv = rsqrtf(vs * (1.f / D_MODEL) + 1e-5f);

  _Float16* orow = out + (size_t)row * D_MODEL;
#pragma unroll
  for (int i = 0; i < D_MODEL / 32; ++i) {
    int c = lane + i * 32;
    orow[c] = (_Float16)((vals[i] - mu) * rinv * g[c] + b[c]);
  }
}

// ----------------------------- WMMA GEMM -----------------------------------
// C[M,N] = A[M,K] * B[N,K]^T (torch Linear convention), fp16 in, fp32 accum.
// Block tile 64x128, BK=32, 256 threads = 8 waves (2 M x 4 N); each wave owns
// a 32x32 tile = 4 v_wmma_f32_16x16x32_f16 accumulators. Tiles are staged
// with GLOBAL_LOAD_ASYNC_TO_LDS_B128 into double-buffered LDS (ASYNCcnt),
// overlapping the next tile fetch with the current WMMAs.
enum {
  EPI_F16 = 0,          // store fp16
  EPI_F32 = 1,          // store fp32
  EPI_SOFTPLUS_F32 = 2, // softplus(c + bias) -> fp32
  EPI_RESID_F32 = 3,    // c + resid -> fp32
  EPI_BIAS_RELU_F16 = 4,// relu(c + bias) -> fp16
  EPI_BIAS_RESID_F32 = 5// c + bias + resid -> fp32
};

template <int EPI>
__global__ __launch_bounds__(256)
void gemm_wmma_f16(const _Float16* __restrict__ A,
                   const _Float16* __restrict__ Bw,
                   float* __restrict__ Cf, _Float16* __restrict__ Ch,
                   const float* __restrict__ bias,
                   const float* __restrict__ resid,
                   int M, int N, int K) {
  constexpr int BM = 64, BN = 128, BK = 32, LDP = BK + 8;
  __shared__ __align__(16) _Float16 As[2][BM][LDP];
  __shared__ __align__(16) _Float16 Bs[2][BN][LDP];

  const int bm = blockIdx.y * BM;
  const int bn = blockIdx.x * BN;
  const int tid  = threadIdx.x;
  const int lane = tid & 31;
  const int wave = tid >> 5;
  const int wm = (wave & 1) * 32;   // wave row-tile inside block (0/32)
  const int wn = (wave >> 1) * 32;  // wave col-tile inside block (0..96)
  const int hi = lane >> 4;         // lane half (K-chunk select)
  const int ml = lane & 15;         // matrix row/col within 16

  // cooperative async-load coordinates
  const int ar = tid >> 2, ac = (tid & 3) * 8;  // A: 64 rows x 4 x b128
  const int br = tid >> 1, bc = (tid & 1) * 16; // B: 128 rows x 2 x (2xb128)
  const bool brow_ok = (bn + br) < N;

  // Pre-zero LDS rows of B that are out of range (K-invariant), both buffers.
  if (N & (BN - 1)) {
    if (!brow_ok) {
      const uint4 z = make_uint4(0u, 0u, 0u, 0u);
#pragma unroll
      for (int buf = 0; buf < 2; ++buf) {
        *(uint4*)&Bs[buf][br][bc]     = z;
        *(uint4*)&Bs[buf][br][bc + 8] = z;
      }
    }
    __syncthreads();
  }

  auto load_tiles_async = [&](int buf, int k0) {
    {  // A tile: one b128 per thread, direct global -> LDS
      const _Float16* gp = A + (size_t)(bm + ar) * K + k0 + ac;
      unsigned lo = (unsigned)(uintptr_t)&As[buf][ar][ac];
      asm volatile("global_load_async_to_lds_b128 %0, %1, off"
                   :: "v"(lo), "v"((unsigned long long)(uintptr_t)gp)
                   : "memory");
    }
    if (brow_ok) {  // B tile: two b128 per thread
      const _Float16* gp = Bw + (size_t)(bn + br) * K + k0 + bc;
      unsigned lo = (unsigned)(uintptr_t)&Bs[buf][br][bc];
      asm volatile("global_load_async_to_lds_b128 %0, %1, off"
                   :: "v"(lo), "v"((unsigned long long)(uintptr_t)gp)
                   : "memory");
      asm volatile("global_load_async_to_lds_b128 %0, %1, off"
                   :: "v"((unsigned)(lo + 16)),
                      "v"((unsigned long long)(uintptr_t)(gp + 8))
                   : "memory");
    }
  };

  v8f acc[2][2] = {};

  const int nk = K / BK;
  load_tiles_async(0, 0);
  asm volatile("s_wait_asynccnt 0x0" ::: "memory");
  __syncthreads();

  for (int i = 0; i < nk; ++i) {
    const int cur = i & 1;
    if (i + 1 < nk) load_tiles_async(cur ^ 1, (i + 1) * BK);

    // A fragments (ISA 16-bit A layout: per-lane K chunks hi*8 and 16+hi*8)
    v16h a[2], b[2];
#pragma unroll
    for (int ti = 0; ti < 2; ++ti) {
      const _Float16* rp = &As[cur][wm + ti * 16 + ml][0];
      ((uint4*)&a[ti])[0] = *(const uint4*)(rp + hi * 8);
      ((uint4*)&a[ti])[1] = *(const uint4*)(rp + 16 + hi * 8);
    }
    // B fragments (same K order as A: per-lane K chunks hi*8 and 16+hi*8)
#pragma unroll
    for (int tj = 0; tj < 2; ++tj) {
      const _Float16* rp = &Bs[cur][wn + tj * 16 + ml][0];
      ((uint4*)&b[tj])[0] = *(const uint4*)(rp + hi * 8);
      ((uint4*)&b[tj])[1] = *(const uint4*)(rp + 16 + hi * 8);
    }
#pragma unroll
    for (int ti = 0; ti < 2; ++ti)
#pragma unroll
      for (int tj = 0; tj < 2; ++tj)
        acc[ti][tj] = __builtin_amdgcn_wmma_f32_16x16x32_f16(
            false, a[ti], false, b[tj], (short)0, acc[ti][tj], false, false);

    asm volatile("s_wait_asynccnt 0x0" ::: "memory");
    __syncthreads();
  }

  // ---- epilogue: D layout -> row = r + hi*8, col = lane&15 ----
#pragma unroll
  for (int ti = 0; ti < 2; ++ti) {
#pragma unroll
    for (int tj = 0; tj < 2; ++tj) {
#pragma unroll
      for (int r = 0; r < 8; ++r) {
        const int row = bm + wm + ti * 16 + r + hi * 8;
        const int col = bn + wn + tj * 16 + ml;
        if (col >= N) continue;
        const float v = acc[ti][tj][r];
        const size_t o = (size_t)row * N + col;
        if constexpr (EPI == EPI_F16) {
          Ch[o] = (_Float16)v;
        } else if constexpr (EPI == EPI_F32) {
          Cf[o] = v;
        } else if constexpr (EPI == EPI_SOFTPLUS_F32) {
          float t = v + bias[col];
          Cf[o] = (t > 20.f) ? t : log1pf(__expf(t));
        } else if constexpr (EPI == EPI_RESID_F32) {
          Cf[o] = v + resid[o];
        } else if constexpr (EPI == EPI_BIAS_RELU_F16) {
          float t = v + bias[col];
          Ch[o] = (_Float16)(t > 0.f ? t : 0.f);
        } else {  // EPI_BIAS_RESID_F32
          Cf[o] = v + bias[col] + resid[o];
        }
      }
    }
  }
}

// --------------------- depthwise causal conv + SiLU ------------------------
// xz is [B, L, 2*D_INNER] fp16; xp half is columns [0, D_INNER).
__global__ void conv_silu_f16(const _Float16* __restrict__ xz,
                              const float* __restrict__ cw,
                              const float* __restrict__ cb,
                              _Float16* __restrict__ xp) {
  int i = blockIdx.x * blockDim.x + threadIdx.x;  // over B*L*D_INNER
  if (i >= BATCH * SEQ * D_INNER) return;
  const int d = i % D_INNER;
  const int t = (i / D_INNER) % SEQ;
  const int b = i / (D_INNER * SEQ);

  float acc = cb[d];
#pragma unroll
  for (int j = 0; j < D_CONV; ++j) {
    int tt = t - (D_CONV - 1) + j;
    if (tt >= 0)
      acc += cw[d * D_CONV + j] *
             (float)xz[((size_t)b * SEQ + tt) * (2 * D_INNER) + d];
  }
  float s = acc / (1.f + __expf(-acc));  // SiLU
  xp[(size_t)i] = (_Float16)s;
}

// ---------------- extract dt-rank slice of x_dbl as fp16 -------------------
__global__ void extract_dtr(const float* __restrict__ xdbl,
                            _Float16* __restrict__ dtr) {
  int i = blockIdx.x * blockDim.x + threadIdx.x;  // over TOKENS*DT_RANK
  if (i >= TOKENS * DT_RANK) return;
  int t = i / DT_RANK, j = i % DT_RANK;
  dtr[i] = (_Float16)xdbl[(size_t)t * XPROJ_N + j];
}

// ------------------- selective-scan + gate, fused --------------------------
// One thread per (batch, channel); sequential over SEQ. D_STATE=2 in regs.
// Emits yg = (y + d_skip*xp) * silu(z) as fp16, ready for the out-proj GEMM.
__global__ void scan_gate_fused(const float* __restrict__ dt,
                                const _Float16* __restrict__ xp,
                                const float* __restrict__ xdbl,
                                const _Float16* __restrict__ xz,
                                const float* __restrict__ a_log,
                                const float* __restrict__ d_skip,
                                _Float16* __restrict__ yg) {
  int i = blockIdx.x * blockDim.x + threadIdx.x;  // over BATCH*D_INNER
  if (i >= BATCH * D_INNER) return;
  const int d = i % D_INNER;
  const int b = i / D_INNER;

  const float A0 = -__expf(a_log[d * D_STATE + 0]);
  const float A1 = -__expf(a_log[d * D_STATE + 1]);
  const float ds = d_skip[d];

  float h0 = 0.f, h1 = 0.f;
  const size_t base = (size_t)b * SEQ;
  for (int t = 0; t < SEQ; ++t) {
    const size_t r = base + t;
    const float dtv = dt[r * D_INNER + d];
    const float xv  = (float)xp[r * D_INNER + d];
    const float* bc = xdbl + r * XPROJ_N + DT_RANK;   // wave-uniform
    const float B0 = bc[0], B1 = bc[1], C0 = bc[2], C1 = bc[3];
    const float zv = (float)xz[r * (2 * D_INNER) + D_INNER + d];

    h0 = __expf(dtv * A0) * h0 + dtv * B0 * xv;
    h1 = __expf(dtv * A1) * h1 + dtv * B1 * xv;
    const float y = h0 * C0 + h1 * C1;
    const float g = zv / (1.f + __expf(-zv));         // silu(z)
    yg[r * D_INNER + d] = (_Float16)((y + ds * xv) * g);
  }
}

// ---------------------------------------------------------------------------
extern "C" void kernel_launch(void* const* d_in, const int* in_sizes, int n_in,
                              void* d_out, int out_size, void* d_ws,
                              size_t ws_size, hipStream_t stream) {
  const float* x       = (const float*)d_in[0];
  const float* w_in    = (const float*)d_in[1];
  const float* conv_w  = (const float*)d_in[2];
  const float* conv_b  = (const float*)d_in[3];
  const float* w_xproj = (const float*)d_in[4];
  const float* w_dt    = (const float*)d_in[5];
  const float* b_dt    = (const float*)d_in[6];
  const float* a_log   = (const float*)d_in[7];
  const float* d_skip  = (const float*)d_in[8];
  const float* w_out   = (const float*)d_in[9];
  const float* ln1_g   = (const float*)d_in[10];
  const float* ln1_b   = (const float*)d_in[11];
  const float* ln2_g   = (const float*)d_in[12];
  const float* ln2_b   = (const float*)d_in[13];
  const float* ffn_w1  = (const float*)d_in[14];
  const float* ffn_b1  = (const float*)d_in[15];
  const float* ffn_w2  = (const float*)d_in[16];
  const float* ffn_b2  = (const float*)d_in[17];
  float* out = (float*)d_out;

  // ---- workspace carve-up (256B aligned) ----
  char* wp = (char*)d_ws;
  auto alloc = [&](size_t bytes) -> void* {
    void* p = (void*)wp;
    wp += (bytes + 255) & ~(size_t)255;
    return p;
  };
  const size_t T = TOKENS;
  _Float16* w_in_h    = (_Float16*)alloc((size_t)2 * D_INNER * D_MODEL * 2);
  _Float16* w_xproj_h = (_Float16*)alloc((size_t)XPROJ_N * D_INNER * 2);
  _Float16* w_dt_h    = (_Float16*)alloc((size_t)D_INNER * DT_RANK * 2);
  _Float16* w_out_h   = (_Float16*)alloc((size_t)D_MODEL * D_INNER * 2);
  _Float16* ffn_w1_h  = (_Float16*)alloc((size_t)D_MODEL * D_MODEL * 2);
  _Float16* ffn_w2_h  = (_Float16*)alloc((size_t)D_MODEL * D_MODEL * 2);
  _Float16* lnx_h     = (_Float16*)alloc(T * D_MODEL * 2);
  _Float16* xz_h      = (_Float16*)alloc(T * 2 * D_INNER * 2);
  _Float16* xp_h      = (_Float16*)alloc(T * D_INNER * 2);
  float*    xdbl_f    = (float*)   alloc(T * XPROJ_N * 4);
  _Float16* dtr_h     = (_Float16*)alloc(T * DT_RANK * 2);
  float*    dt_f      = (float*)   alloc(T * D_INNER * 4);
  _Float16* yg_h      = (_Float16*)alloc(T * D_INNER * 2);
  float*    h_f       = (float*)   alloc(T * D_MODEL * 4);
  _Float16* f2_h      = (_Float16*)alloc(T * D_MODEL * 2);
  _Float16* t1_h      = lnx_h;  // lnx dead after GEMM1 -> reuse for FFN hidden

  // ---- weight downconversion (fp32 -> fp16) ----
  auto cvt = [&](const float* src, _Float16* dst, int n) {
    cvt_f32_to_f16<<<(n + 255) / 256, 256, 0, stream>>>(src, dst, n);
  };
  cvt(w_in,    w_in_h,    2 * D_INNER * D_MODEL);
  cvt(w_xproj, w_xproj_h, XPROJ_N * D_INNER);
  cvt(w_dt,    w_dt_h,    D_INNER * DT_RANK);
  cvt(w_out,   w_out_h,   D_MODEL * D_INNER);
  cvt(ffn_w1,  ffn_w1_h,  D_MODEL * D_MODEL);
  cvt(ffn_w2,  ffn_w2_h,  D_MODEL * D_MODEL);

  auto gemm_grid = [](int M, int N) {
    return dim3((N + 127) / 128, (M + 63) / 64);
  };

  // 1) LN1
  layernorm_f16<<<TOKENS / 4, 128, 0, stream>>>(x, ln1_g, ln1_b, lnx_h);
  // 2) xz = ln_x @ w_in^T  -> fp16
  gemm_wmma_f16<EPI_F16><<<gemm_grid(T, 2 * D_INNER), 256, 0, stream>>>(
      lnx_h, w_in_h, nullptr, xz_h, nullptr, nullptr, T, 2 * D_INNER, D_MODEL);
  // 3) depthwise causal conv + SiLU -> xp fp16
  conv_silu_f16<<<(BATCH * SEQ * D_INNER) / 256, 256, 0, stream>>>(
      xz_h, conv_w, conv_b, xp_h);
  // 4) x_dbl = xp @ w_xproj^T -> fp32 (N=68, zero-padded rows)
  gemm_wmma_f16<EPI_F32><<<gemm_grid(T, XPROJ_N), 256, 0, stream>>>(
      xp_h, w_xproj_h, xdbl_f, nullptr, nullptr, nullptr, T, XPROJ_N, D_INNER);
  // 5) dt-rank slice -> fp16
  extract_dtr<<<(TOKENS * DT_RANK) / 256, 256, 0, stream>>>(xdbl_f, dtr_h);
  // 6) dt = softplus(dtr @ w_dt^T + b_dt) -> fp32
  gemm_wmma_f16<EPI_SOFTPLUS_F32><<<gemm_grid(T, D_INNER), 256, 0, stream>>>(
      dtr_h, w_dt_h, dt_f, nullptr, b_dt, nullptr, T, D_INNER, DT_RANK);
  // 7) selective scan + D-skip + SiLU(z) gate -> yg fp16
  scan_gate_fused<<<(BATCH * D_INNER) / 256, 256, 0, stream>>>(
      dt_f, xp_h, xdbl_f, xz_h, a_log, d_skip, yg_h);
  // 8) h = x + yg @ w_out^T -> fp32
  gemm_wmma_f16<EPI_RESID_F32><<<gemm_grid(T, D_MODEL), 256, 0, stream>>>(
      yg_h, w_out_h, h_f, nullptr, nullptr, x, T, D_MODEL, D_INNER);
  // 9) LN2
  layernorm_f16<<<TOKENS / 4, 128, 0, stream>>>(h_f, ln2_g, ln2_b, f2_h);
  // 10) t1 = relu(f @ ffn_w1^T + b1) -> fp16
  gemm_wmma_f16<EPI_BIAS_RELU_F16><<<gemm_grid(T, D_MODEL), 256, 0, stream>>>(
      f2_h, ffn_w1_h, nullptr, t1_h, ffn_b1, nullptr, T, D_MODEL, D_MODEL);
  // 11) out = h + t1 @ ffn_w2^T + b2 -> fp32
  gemm_wmma_f16<EPI_BIAS_RESID_F32><<<gemm_grid(T, D_MODEL), 256, 0, stream>>>(
      t1_h, ffn_w2_h, out, nullptr, ffn_b2, h_f, T, D_MODEL, D_MODEL);

  (void)in_sizes; (void)n_in; (void)out_size; (void)ws_size;
}
